// SuperGlue_8881992368456
// MI455X (gfx1250) — compile-verified, matrix operand order checked by reading, not run on hardware
//
#include <hip/hip_runtime.h>
#include <hip/hip_bf16.h>
#include <math.h>

// ---------------------------------------------------------------------------
// SuperGlue forward for MI455X (gfx1250, wave32, WMMA).
// Dense GEMMs: v_wmma_f32_16x16x32_bf16 (bf16 in, fp32 accum), with LDS tiles
// stored pre-swizzled into WMMA fragment order so each lane loads its whole
// fragment with two ds_load_b128 ops.
// ---------------------------------------------------------------------------

#define BB 2
#define NN 1024
#define DD 256
#define HH 4
#define HDIM 64
#define LL 18

typedef __attribute__((ext_vector_type(16))) __bf16 v16bf;
typedef __attribute__((ext_vector_type(8)))  __bf16 v8bf;
typedef __attribute__((ext_vector_type(8)))  float  v8f;

static __device__ __forceinline__ unsigned short f2bf_bits(float f) {
    unsigned int u = __builtin_bit_cast(unsigned int, f);
    unsigned int r = u + 0x7FFFu + ((u >> 16) & 1u);   // round-to-nearest-even
    return (unsigned short)(r >> 16);
}
static __device__ __forceinline__ __bf16 f2bf(float f) {
    unsigned short b = f2bf_bits(f);
    return __builtin_bit_cast(__bf16, b);
}

// Inverse of the documented 16-bit A/B fragment layout:
// for K index k in [0,32): which half-lane group (0: lanes 0-15, 1: lanes 16-31)
// and which packed half slot h in [0,16) holds it.
static __device__ __forceinline__ void frag_pos(int k, int& hib, int& h) {
    hib = (k >> 3) & 1;                 // K 8-15 / 24-31 live in lanes 16-31
    int klow = k - 8 * hib;             // in {0..7} U {16..23}
    int r = (klow < 8) ? (klow >> 1) : (4 + ((klow - 16) >> 1));
    h = 2 * r + (k & 1);
}

static __device__ __forceinline__ v16bf load_frag(const __bf16* p) {
    const v8bf* vp = (const v8bf*)p;
    v8bf a = vp[0], b = vp[1];          // two ds_load_b128
    return __builtin_shufflevector(a, b, 0,1,2,3,4,5,6,7,8,9,10,11,12,13,14,15);
}

// ---------------------------------------------------------------------------
// Generic GEMM:  Y[z][o,n] = sum_k W[o,k] X[z][k,n] + bias[o] (+Yold if ACCUM)
// W row-major with leading dim Wld. 64x64 tile per 256-thread block,
// 8 waves x (1 A frag, 2 B frags, 2 WMMA) per 32-deep K step.
// ---------------------------------------------------------------------------
template <int ACCUM>
__global__ __launch_bounds__(256) void gemm_wmma(
    const float* __restrict__ W, int Wld,
    const float* __restrict__ X,
    const float* __restrict__ bias,
    float* __restrict__ Y,
    int Ci, int Ncols)
{
    const int z  = blockIdx.z;
    const int n0 = blockIdx.x * 64;
    const int o0 = blockIdx.y * 64;
    const float* Xz = X + (size_t)z * Ci * Ncols;
    const int Co = gridDim.y * 64;
    float* Yz = Y + (size_t)z * Co * Ncols;

    // Fragment-order tiles: [16-row tile][lane][half]
    __shared__ __align__(32) __bf16 Afr[4][32][16];
    __shared__ __align__(32) __bf16 Bfr[4][32][16];

    const int tid  = threadIdx.x;
    const int lane = tid & 31;
    const int wv   = tid >> 5;
    const int trow = wv & 3;        // 16-row group of o
    const int tcp  = wv >> 2;       // 32-col group of n
    const int l16  = lane & 15;
    const int hi   = (lane >> 4) & 1;

    v8f acc0 = {}; v8f acc1 = {};

    for (int kk = 0; kk < Ci; kk += 32) {
        #pragma unroll
        for (int j = 0; j < 8; ++j) {
            int idx = tid + j * 256;
            int o = idx >> 5, k = idx & 31;
            int hib, h; frag_pos(k, hib, h);
            Afr[o >> 4][(o & 15) + 16 * hib][h] =
                f2bf(W[(size_t)(o0 + o) * Wld + kk + k]);
        }
        #pragma unroll
        for (int j = 0; j < 8; ++j) {
            int idx = tid + j * 256;
            int k = idx >> 6, c = idx & 63;
            int hib, h; frag_pos(k, hib, h);
            Bfr[c >> 4][(c & 15) + 16 * hib][h] =
                f2bf(Xz[(size_t)(kk + k) * Ncols + n0 + c]);
        }
        if (kk + 32 < Ci) {   // prefetch next K tile (global_prefetch_b8)
            __builtin_prefetch(&W[(size_t)(o0 + (tid >> 2)) * Wld + kk + 32], 0, 1);
            __builtin_prefetch(&Xz[(size_t)(kk + 32 + (tid >> 4)) * Ncols + n0 + (tid & 15) * 4], 0, 1);
        }
        __syncthreads();

        v16bf af  = load_frag(Afr[trow][lane]);
        v16bf bf0 = load_frag(Bfr[tcp * 2][lane]);
        v16bf bf1 = load_frag(Bfr[tcp * 2 + 1][lane]);
        acc0 = __builtin_amdgcn_wmma_f32_16x16x32_bf16(false, af, false, bf0,
                                                       (short)0, acc0, false, false);
        acc1 = __builtin_amdgcn_wmma_f32_16x16x32_bf16(false, af, false, bf1,
                                                       (short)0, acc1, false, false);
        __syncthreads();
    }

    #pragma unroll
    for (int r = 0; r < 8; ++r) {
        int o = o0 + trow * 16 + r + hi * 8;
        float bv = bias[o];
        int na = n0 + tcp * 32 + l16;
        size_t ia = (size_t)o * Ncols + na;
        size_t ib = ia + 16;
        float va = acc0[r] + bv;
        float vb = acc1[r] + bv;
        if (ACCUM) { va += Yz[ia]; vb += Yz[ib]; }
        Yz[ia] = va; Yz[ib] = vb;
    }
}

// ---------------------------------------------------------------------------
// Attention scores: S[bh][n,m] = scale * sum_dd Q[b][dd*H+h][n] * K[b][dd*H+h][m]
// (also used for final m0^T m1 with H=1, Kdim=256)
// ---------------------------------------------------------------------------
__global__ __launch_bounds__(256) void attn_scores(
    const float* __restrict__ Q, const float* __restrict__ K,
    float* __restrict__ S, int N, int H, int Kdim, float scale)
{
    const int bh = blockIdx.z, b = bh / H, h = bh % H;
    const int Dst = H * Kdim;
    const float* Qb = Q + (size_t)b * Dst * N;
    const float* Kb = K + (size_t)b * Dst * N;
    float* Sb = S + (size_t)bh * N * N;
    const int m0 = blockIdx.x * 64, n0 = blockIdx.y * 64;

    __shared__ __align__(32) __bf16 QA[4][32][16];   // A = Q^T fragments
    __shared__ __align__(32) __bf16 KB[4][32][16];   // B = K fragments

    const int tid = threadIdx.x, lane = tid & 31, wv = tid >> 5;
    const int trow = wv & 3, tcp = wv >> 2;
    const int l16 = lane & 15, hi = (lane >> 4) & 1;

    v8f acc0 = {}; v8f acc1 = {};

    for (int kk = 0; kk < Kdim; kk += 32) {
        #pragma unroll
        for (int j = 0; j < 8; ++j) {
            int idx = tid + j * 256;
            int dk = idx >> 6, c = idx & 63;
            int hib, hh; frag_pos(dk, hib, hh);
            float qv = Qb[(size_t)((kk + dk) * H + h) * N + n0 + c];
            float kv = Kb[(size_t)((kk + dk) * H + h) * N + m0 + c];
            QA[c >> 4][(c & 15) + 16 * hib][hh] = f2bf(qv);
            KB[c >> 4][(c & 15) + 16 * hib][hh] = f2bf(kv);
        }
        __syncthreads();

        v16bf af  = load_frag(QA[trow][lane]);
        v16bf bf0 = load_frag(KB[tcp * 2][lane]);
        v16bf bf1 = load_frag(KB[tcp * 2 + 1][lane]);
        acc0 = __builtin_amdgcn_wmma_f32_16x16x32_bf16(false, af, false, bf0,
                                                       (short)0, acc0, false, false);
        acc1 = __builtin_amdgcn_wmma_f32_16x16x32_bf16(false, af, false, bf1,
                                                       (short)0, acc1, false, false);
        __syncthreads();
    }

    #pragma unroll
    for (int r = 0; r < 8; ++r) {
        int n = n0 + trow * 16 + r + hi * 8;
        size_t ia = (size_t)n * N + m0 + tcp * 32 + l16;
        Sb[ia]      = scale * acc0[r];
        Sb[ia + 16] = scale * acc1[r];
    }
}

// ---------------------------------------------------------------------------
// msg: Out[b][dd*H+h][n] = sum_m V[b][dd*H+h][m] * P[bh][n][m]
// ---------------------------------------------------------------------------
__global__ __launch_bounds__(256) void attn_apply(
    const float* __restrict__ P, const float* __restrict__ V,
    float* __restrict__ Out, int N, int H, int Kdim)
{
    const int bh = blockIdx.z, b = bh / H, h = bh % H;
    const int Dst = H * Kdim;
    const int n0 = blockIdx.x * 64, dd0 = blockIdx.y * 64;
    const float* Pb = P + (size_t)bh * N * N;
    const float* Vb = V + (size_t)b * Dst * N;
    float* Ob = Out + (size_t)b * Dst * N;

    __shared__ __align__(32) __bf16 VA[4][32][16];   // A = V fragments
    __shared__ __align__(32) __bf16 PB[4][32][16];   // B = P^T fragments

    const int tid = threadIdx.x, lane = tid & 31, wv = tid >> 5;
    const int trow = wv & 3, tcp = wv >> 2;
    const int l16 = lane & 15, hi = (lane >> 4) & 1;

    v8f acc0 = {}; v8f acc1 = {};

    for (int mm = 0; mm < N; mm += 32) {
        #pragma unroll
        for (int j = 0; j < 8; ++j) {
            int idx = tid + j * 256;
            int r = idx >> 5, c = idx & 31;          // r = dd row, c = k (m)
            int hib, hh; frag_pos(c, hib, hh);
            VA[r >> 4][(r & 15) + 16 * hib][hh] =
                f2bf(Vb[(size_t)((dd0 + r) * H + h) * N + mm + c]);
        }
        #pragma unroll
        for (int j = 0; j < 8; ++j) {
            int idx = tid + j * 256;
            int mk = idx & 31, n = idx >> 5;         // k = mk, col = n
            int hib, hh; frag_pos(mk, hib, hh);
            PB[n >> 4][(n & 15) + 16 * hib][hh] =
                f2bf(Pb[(size_t)(n0 + n) * N + mm + mk]);
        }
        __syncthreads();

        v16bf af  = load_frag(VA[trow][lane]);
        v16bf bf0 = load_frag(PB[tcp * 2][lane]);
        v16bf bf1 = load_frag(PB[tcp * 2 + 1][lane]);
        acc0 = __builtin_amdgcn_wmma_f32_16x16x32_bf16(false, af, false, bf0,
                                                       (short)0, acc0, false, false);
        acc1 = __builtin_amdgcn_wmma_f32_16x16x32_bf16(false, af, false, bf1,
                                                       (short)0, acc1, false, false);
        __syncthreads();
    }

    #pragma unroll
    for (int r = 0; r < 8; ++r) {
        int dd = dd0 + trow * 16 + r + hi * 8;
        size_t ia = (size_t)(dd * H + h) * N + n0 + tcp * 32 + l16;
        Ob[ia]      = acc0[r];
        Ob[ia + 16] = acc1[r];
    }
}

// ---------------------------------------------------------------------------
// Row softmax over N (in place).
// ---------------------------------------------------------------------------
__global__ __launch_bounds__(256) void softmax_rows(float* __restrict__ P, int N)
{
    __shared__ float red[256];
    float* p = P + (size_t)blockIdx.x * N;
    const int t = threadIdx.x;

    float m = -3.4e38f;
    for (int i = t; i < N; i += 256) m = fmaxf(m, p[i]);
    red[t] = m; __syncthreads();
    for (int s = 128; s > 0; s >>= 1) { if (t < s) red[t] = fmaxf(red[t], red[t+s]); __syncthreads(); }
    const float mx = red[0]; __syncthreads();

    float sum = 0.0f;
    for (int i = t; i < N; i += 256) sum += expf(p[i] - mx);
    red[t] = sum; __syncthreads();
    for (int s = 128; s > 0; s >>= 1) { if (t < s) red[t] += red[t+s]; __syncthreads(); }
    const float inv = 1.0f / red[0]; __syncthreads();

    for (int i = t; i < N; i += 256) p[i] = expf(p[i] - mx) * inv;
}

// ---------------------------------------------------------------------------
// Channel LayerNorm (unbiased std, eps added to std) + ReLU, per column (b,n).
// X: [Bc, C, N], in place. C multiple of 256 (here 512).
// ---------------------------------------------------------------------------
__global__ __launch_bounds__(256) void ln_relu_cols(
    float* __restrict__ X, const float* __restrict__ g,
    const float* __restrict__ be, int C, int N)
{
    __shared__ float red[256];
    const int col = blockIdx.x, b = col / N, n = col % N;
    float* base = X + (size_t)b * C * N + n;
    const int t = threadIdx.x;
    const int per = C / 256;

    float vals[4];
    float s = 0.0f;
    for (int i = 0; i < per; ++i) { vals[i] = base[(size_t)(t + i * 256) * N]; s += vals[i]; }
    red[t] = s; __syncthreads();
    for (int ss = 128; ss > 0; ss >>= 1) { if (t < ss) red[t] += red[t+ss]; __syncthreads(); }
    const float mean = red[0] / (float)C; __syncthreads();

    float q = 0.0f;
    for (int i = 0; i < per; ++i) { float d = vals[i] - mean; q += d * d; }
    red[t] = q; __syncthreads();
    for (int ss = 128; ss > 0; ss >>= 1) { if (t < ss) red[t] += red[t+ss]; __syncthreads(); }
    const float stdv = sqrtf(red[0] / (float)(C - 1)); __syncthreads();

    const float inv = 1.0f / (stdv + 1e-6f);
    for (int i = 0; i < per; ++i) {
        int c = t + i * 256;
        float o = g[c] * (vals[i] - mean) * inv + be[c];
        base[(size_t)c * N] = fmaxf(o, 0.0f);
    }
}

// ---------------------------------------------------------------------------
// Keypoint encoder: per-column MLP 3->32->64->128->256 with channel-LN + ReLU,
// output = desc + kenc. One 128-thread block per (stream, b, n).
// ---------------------------------------------------------------------------
static __device__ float ln_relu_val(float y, int C, int t, float* red,
                                    const float* g, const float* be)
{
    const bool act = (t < C);
    red[t] = act ? y : 0.0f; __syncthreads();
    for (int s = 64; s > 0; s >>= 1) { if (t < s) red[t] += red[t+s]; __syncthreads(); }
    const float mean = red[0] / (float)C; __syncthreads();
    float d = act ? (y - mean) : 0.0f;
    red[t] = d * d; __syncthreads();
    for (int s = 64; s > 0; s >>= 1) { if (t < s) red[t] += red[t+s]; __syncthreads(); }
    const float stdv = sqrtf(red[0] / (float)(C - 1)); __syncthreads();
    float o = 0.0f;
    if (act) {
        o = g[t] * (y - mean) / (stdv + 1e-6f) + be[t];
        o = fmaxf(o, 0.0f);
    }
    return o;
}

__global__ __launch_bounds__(128) void kenc_kernel(
    const float* __restrict__ desc0, const float* __restrict__ desc1,
    const float* __restrict__ kpts0, const float* __restrict__ kpts1,
    const float* __restrict__ sc0,   const float* __restrict__ sc1,
    const float* __restrict__ w0, const float* __restrict__ b0,
    const float* __restrict__ g0, const float* __restrict__ e0,
    const float* __restrict__ w1, const float* __restrict__ b1,
    const float* __restrict__ g1, const float* __restrict__ e1,
    const float* __restrict__ w2, const float* __restrict__ b2,
    const float* __restrict__ g2, const float* __restrict__ e2,
    const float* __restrict__ w3, const float* __restrict__ b3,
    float* __restrict__ out0, float* __restrict__ out1)
{
    const int gid = blockIdx.x;
    const int s   = gid / (BB * NN);
    const int rem = gid % (BB * NN);
    const int b = rem / NN, n = rem % NN;
    const float* kp  = (s ? kpts1 : kpts0) + (size_t)(b * NN + n) * 2;
    const float  in2 = (s ? sc1 : sc0)[b * NN + n];
    const float* dsc = s ? desc1 : desc0;
    float* out = s ? out1 : out0;

    __shared__ float xb[128];
    __shared__ float red[128];
    const int t = threadIdx.x;
    const float in0 = kp[0], in1 = kp[1];

    // layer 0: 3 -> 32
    float y = 0.0f;
    if (t < 32) y = w0[t*3+0]*in0 + w0[t*3+1]*in1 + w0[t*3+2]*in2 + b0[t];
    float r0 = ln_relu_val(y, 32, t, red, g0, e0);
    __syncthreads(); if (t < 32) xb[t] = r0; __syncthreads();

    // layer 1: 32 -> 64
    y = 0.0f;
    if (t < 64) { float a = b1[t]; for (int i = 0; i < 32; ++i) a += w1[t*32+i]*xb[i]; y = a; }
    float r1 = ln_relu_val(y, 64, t, red, g1, e1);
    __syncthreads(); if (t < 64) xb[t] = r1; __syncthreads();

    // layer 2: 64 -> 128
    { float a = b2[t]; for (int i = 0; i < 64; ++i) a += w2[t*64+i]*xb[i]; y = a; }
    float r2 = ln_relu_val(y, 128, t, red, g2, e2);
    __syncthreads(); xb[t] = r2; __syncthreads();

    // layer 3: 128 -> 256 (two channels per thread), + desc residual
    float ya = b3[t], yb = b3[t + 128];
    for (int i = 0; i < 128; ++i) {
        float xv = xb[i];
        ya += w3[(size_t)t * 128 + i] * xv;
        yb += w3[(size_t)(t + 128) * 128 + i] * xv;
    }
    out[((size_t)b * DD + t)       * NN + n] = dsc[((size_t)b * DD + t)       * NN + n] + ya;
    out[((size_t)b * DD + t + 128) * NN + n] = dsc[((size_t)b * DD + t + 128) * NN + n] + yb;
}

// ---------------------------------------------------------------------------
// Sinkhorn kernels (stable logsumexp).
// ---------------------------------------------------------------------------
__global__ __launch_bounds__(256) void sink_row(
    const float* __restrict__ sc, const float* __restrict__ vv,
    float* __restrict__ uu, int N, float norm)
{
    __shared__ float red[256];
    const int row = blockIdx.x, b = row / N;
    const float* p  = sc + (size_t)row * N;
    const float* vb = vv + (size_t)b * N;
    const int t = threadIdx.x;

    float m = -3.4e38f;
    for (int i = t; i < N; i += 256) m = fmaxf(m, p[i] + vb[i]);
    red[t] = m; __syncthreads();
    for (int s = 128; s > 0; s >>= 1) { if (t < s) red[t] = fmaxf(red[t], red[t+s]); __syncthreads(); }
    const float mx = red[0]; __syncthreads();

    float sum = 0.0f;
    for (int i = t; i < N; i += 256) sum += expf(p[i] + vb[i] - mx);
    red[t] = sum; __syncthreads();
    for (int s = 128; s > 0; s >>= 1) { if (t < s) red[t] += red[t+s]; __syncthreads(); }
    if (t == 0) uu[row] = norm - (mx + logf(red[0]));
}

__global__ __launch_bounds__(256) void sink_col(
    const float* __restrict__ sc, const float* __restrict__ uu,
    float* __restrict__ vv, int N, float norm)
{
    __shared__ float red[256];
    const int cm = blockIdx.x, b = cm / N, mcol = cm % N;
    const float* base = sc + (size_t)b * N * N + mcol;
    const float* ub = uu + (size_t)b * N;
    const int t = threadIdx.x;

    float m = -3.4e38f;
    for (int i = t; i < N; i += 256) m = fmaxf(m, base[(size_t)i * N] + ub[i]);
    red[t] = m; __syncthreads();
    for (int s = 128; s > 0; s >>= 1) { if (t < s) red[t] = fmaxf(red[t], red[t+s]); __syncthreads(); }
    const float mx = red[0]; __syncthreads();

    float sum = 0.0f;
    for (int i = t; i < N; i += 256) sum += expf(base[(size_t)i * N] + ub[i] - mx);
    red[t] = sum; __syncthreads();
    for (int s = 128; s > 0; s >>= 1) { if (t < s) red[t] += red[t+s]; __syncthreads(); }
    if (t == 0) vv[cm] = norm - (mx + logf(red[0]));
}

__global__ __launch_bounds__(256) void init_zero(float* __restrict__ p, int n)
{
    int i = blockIdx.x * 256 + threadIdx.x;
    if (i < n) p[i] = 0.0f;
}

__global__ __launch_bounds__(256) void final_z(
    const float* __restrict__ sc, const float* __restrict__ uu,
    const float* __restrict__ vv, float* __restrict__ out, int N, float norm)
{
    size_t i = (size_t)blockIdx.x * 256 + threadIdx.x;
    int mcol = (int)(i % N);
    size_t r = i / N;
    int n = (int)(r % N);
    int b = (int)(r / N);
    out[i] = sc[i] + uu[b * N + n] + vv[b * N + mcol] - norm;
}

// ---------------------------------------------------------------------------
// Host-side orchestration.
// ---------------------------------------------------------------------------
extern "C" void kernel_launch(void* const* d_in, const int* in_sizes, int n_in,
                              void* d_out, int out_size, void* d_ws, size_t ws_size,
                              hipStream_t stream)
{
    const float* desc0 = (const float*)d_in[0];
    const float* desc1 = (const float*)d_in[1];
    const float* kpts0 = (const float*)d_in[2];
    const float* kpts1 = (const float*)d_in[3];
    const float* scr0  = (const float*)d_in[4];
    const float* scr1  = (const float*)d_in[5];
    const float* kw0 = (const float*)d_in[6],  *kb0 = (const float*)d_in[7];
    const float* kg0 = (const float*)d_in[8],  *ke0 = (const float*)d_in[9];
    const float* kw1 = (const float*)d_in[10], *kb1 = (const float*)d_in[11];
    const float* kg1 = (const float*)d_in[12], *ke1 = (const float*)d_in[13];
    const float* kw2 = (const float*)d_in[14], *kb2 = (const float*)d_in[15];
    const float* kg2 = (const float*)d_in[16], *ke2 = (const float*)d_in[17];
    const float* kw3 = (const float*)d_in[18], *kb3 = (const float*)d_in[19];
    const float* gqw = (const float*)d_in[20], *gqb = (const float*)d_in[21];
    const float* gkw = (const float*)d_in[22], *gkb = (const float*)d_in[23];
    const float* gvw = (const float*)d_in[24], *gvb = (const float*)d_in[25];
    const float* gmw = (const float*)d_in[26], *gmb = (const float*)d_in[27];
    const float* gw1 = (const float*)d_in[28], *gb1 = (const float*)d_in[29];
    const float* gg1 = (const float*)d_in[30], *ge1 = (const float*)d_in[31];
    const float* gw2 = (const float*)d_in[32], *gb2 = (const float*)d_in[33];
    const float* fw  = (const float*)d_in[34], *fb  = (const float*)d_in[35];
    float* out = (float*)d_out;

    const int B = BB, N = NN, D = DD, H = HH, HD = HDIM, L = LL;
    const size_t tens = (size_t)B * D * N;

    float* ws = (float*)d_ws;
    size_t off = 0;
    auto alloc = [&](size_t nflt) { float* p = ws + off; off += nflt; return p; };
    float* dbuf0 = alloc(tens);
    float* dbuf1 = alloc(tens);
    float* qbuf  = alloc(tens);
    float* kbuf  = alloc(tens);
    float* vbuf  = alloc(tens);
    float* msgb  = alloc(tens);
    float* mpb   = alloc(tens);
    float* y1b0  = alloc(2 * tens);
    float* y1b1  = alloc(2 * tens);
    float* uu    = alloc((size_t)B * N);
    float* vv    = alloc((size_t)B * N);
    float* zbias = alloc(2 * D);                   // zero bias stub
    float* prob  = alloc((size_t)B * H * N * N);   // reused for final scores
    (void)ws_size; (void)n_in; (void)in_sizes; (void)out_size;

    // zero u, v and the bias stub (workspace is poisoned by the harness)
    init_zero<<<dim3((2 * B * N + 2 * D + 255) / 256), 256, 0, stream>>>(
        uu, 2 * B * N + 2 * D);   // uu, vv, zbias are contiguous

    // 1) keypoint encoder + residual
    kenc_kernel<<<dim3(2 * B * N), 128, 0, stream>>>(
        desc0, desc1, kpts0, kpts1, scr0, scr1,
        kw0, kb0, kg0, ke0, kw1, kb1, kg1, ke1, kw2, kb2, kg2, ke2, kw3, kb3,
        dbuf0, dbuf1);

    const dim3 gD (N / 64, D / 64, B);        // Co=256 GEMMs
    const dim3 g2D(N / 64, 2 * D / 64, B);    // Co=512 GEMMs
    const dim3 gSC(N / 64, N / 64, B * H);    // attention scores
    const dim3 gAP(N / 64, HD / 64, B * H);   // attention apply

    // 2) GNN layers
    for (int i = 0; i < L; ++i) {
        const float* qw = gqw + (size_t)i * D * D;
        const float* kw = gkw + (size_t)i * D * D;
        const float* vw = gvw + (size_t)i * D * D;
        const float* mw = gmw + (size_t)i * D * D;
        const float* w1 = gw1 + (size_t)i * 4 * D * D;     // [2D, 2D]
        const float* w2 = gw2 + (size_t)i * 2 * D * D;     // [D, 2D]
        const float* qb = gqb + (size_t)i * D;
        const float* kb = gkb + (size_t)i * D;
        const float* vb = gvb + (size_t)i * D;
        const float* mb = gmb + (size_t)i * D;
        const float* b1 = gb1 + (size_t)i * 2 * D;
        const float* g1 = gg1 + (size_t)i * 2 * D;
        const float* e1 = ge1 + (size_t)i * 2 * D;
        const float* b2 = gb2 + (size_t)i * D;

        for (int s = 0; s < 2; ++s) {
            float* x     = s ? dbuf1 : dbuf0;
            float* other = s ? dbuf0 : dbuf1;
            float* src   = (i & 1) ? other : x;    // odd layers: cross
            float* y1    = s ? y1b1 : y1b0;

            gemm_wmma<0><<<gD, 256, 0, stream>>>(qw, D, x,   qb, qbuf, D, N);
            gemm_wmma<0><<<gD, 256, 0, stream>>>(kw, D, src, kb, kbuf, D, N);
            gemm_wmma<0><<<gD, 256, 0, stream>>>(vw, D, src, vb, vbuf, D, N);

            attn_scores<<<gSC, 256, 0, stream>>>(qbuf, kbuf, prob, N, H, HD, 0.125f);
            softmax_rows<<<dim3(B * H * N), 256, 0, stream>>>(prob, N);
            attn_apply<<<gAP, 256, 0, stream>>>(prob, vbuf, msgb, N, H, HD);

            gemm_wmma<0><<<gD, 256, 0, stream>>>(mw, D, msgb, mb, mpb, D, N);

            // y1 = W1[:, :D] x + b1 ; y1 += W1[:, D:] msg_proj ; LN + ReLU
            gemm_wmma<0><<<g2D, 256, 0, stream>>>(w1,     2 * D, x,   b1,    y1, D, N);
            gemm_wmma<1><<<g2D, 256, 0, stream>>>(w1 + D, 2 * D, mpb, zbias, y1, D, N);
            ln_relu_cols<<<dim3(B * N), 256, 0, stream>>>(y1, g1, e1, 2 * D, N);
        }
        // residual updates last (both streams' k/v already consumed old d's)
        gemm_wmma<1><<<gD, 256, 0, stream>>>(w2, 2 * D, y1b0, b2, dbuf0, 2 * D, N);
        gemm_wmma<1><<<gD, 256, 0, stream>>>(w2, 2 * D, y1b1, b2, dbuf1, 2 * D, N);
    }

    // 3) final projection + score matrix (reuse qbuf/kbuf as m0/m1, prob as sc)
    gemm_wmma<0><<<gD, 256, 0, stream>>>(fw, D, dbuf0, fb, qbuf, D, N);
    gemm_wmma<0><<<gD, 256, 0, stream>>>(fw, D, dbuf1, fb, kbuf, D, N);
    float* scfin = prob;
    attn_scores<<<dim3(N / 64, N / 64, B), 256, 0, stream>>>(
        qbuf, kbuf, scfin, N, 1, D, 1.0f / 16.0f);

    // 4) Sinkhorn, 100 iterations, uniform log-marginals
    const float norm = -logf(2.0f * (float)N);
    for (int it = 0; it < 100; ++it) {
        sink_row<<<dim3(B * N), 256, 0, stream>>>(scfin, vv, uu, N, norm);
        sink_col<<<dim3(B * N), 256, 0, stream>>>(scfin, uu, vv, N, norm);
    }

    // 5) Z = sc + u + v - norm
    final_z<<<dim3((B * N * N) / 256), 256, 0, stream>>>(scfin, uu, vv, out, N, norm);
}
